// Causal_model_vae_37254546325831
// MI455X (gfx1250) — matrix operand order repaired and checked by reading, not compiled, not run on hardware
//
#include <hip/hip_runtime.h>

// ---------------------------------------------------------------------------
// Problem constants (from the reference)
// ---------------------------------------------------------------------------
#define PN  16384
#define PDX 1024
#define PDH 1024
#define PS  8

// GEMM tiling: block = 256 threads = 8 wave32, block tile 128x128, K-step 32.
// Waves in 2x4 grid; each wave owns a 64x32 sub-tile = 4x2 WMMA fragments.
#define TM  128
#define TNB 128
#define TK  32
#define CAP_ROWS (PN + PS * TM)   // 17408: worst-case padded compact rows

typedef __attribute__((ext_vector_type(16))) __bf16 v16bf;
typedef __attribute__((ext_vector_type(8)))  __bf16 v8bf;
typedef __attribute__((ext_vector_type(8)))  float  v8f;

// ---------------------------------------------------------------------------
// CDNA5 async global->LDS copy (ASYNCcnt). GVS addressing:
//   mem = SGPR64 + VGPR32_offset ; lds = VGPR32 lds byte addr
// ---------------------------------------------------------------------------
__device__ __forceinline__ void async_cp_b128(unsigned lds_off, unsigned gvoff,
                                              unsigned long sbase) {
    asm volatile("global_load_async_to_lds_b128 %0, %1, %2"
                 :: "v"(lds_off), "v"(gvoff), "s"(sbase)
                 : "memory");
}

// ---------------------------------------------------------------------------
// CDNA5 Tensor Data Mover: one instruction DMAs a whole 2D tile to LDS
// (TENSORcnt). D# bit layout per CDNA5 ISA.
// This toolchain exposes the 6-arg builtin:
//   (uint32x4 g0, int32x8 g1, int32x4 g2, int32x4 g3, int32x8 pad, i32 cpol)
// ---------------------------------------------------------------------------
#if defined(__has_builtin)
#if __has_builtin(__builtin_amdgcn_tensor_load_to_lds)
#define HAVE_TDM 1
#endif
#endif
#ifndef HAVE_TDM
#define HAVE_TDM 0
#endif

#if HAVE_TDM
typedef __attribute__((ext_vector_type(4))) unsigned tdm_v4u;
typedef __attribute__((ext_vector_type(8))) int      tdm_v8i;
typedef __attribute__((ext_vector_type(4))) int      tdm_v4i;

// Load tile_rows x TK (bf16) tile, row stride = stride_elems, into LDS.
__device__ __forceinline__ void tdm_load_2d(unsigned lds_addr, unsigned long gaddr,
                                            int tile_rows, int stride_elems) {
    tdm_v4u g0;
    g0[0] = 1u;                                             // count=1, user D#
    g0[1] = lds_addr;                                       // LDS byte address
    g0[2] = (unsigned)(gaddr & 0xffffffffu);                // global addr lo
    g0[3] = (unsigned)((gaddr >> 32) & 0x01ffffffu)         // global addr hi
          | (2u << 30);                                     // type = 2 (image)
    tdm_v8i g1;
    g1[0] = 1 << 16;                                        // data_size = 2B
    g1[1] = (int)((unsigned)(stride_elems & 0xffff) << 16); // tensor_dim0 lo16
    g1[2] = (int)(((unsigned)stride_elems >> 16) & 0xffffu) // tensor_dim0 hi16
          | (int)(0xffffu << 16);                           // tensor_dim1 lo16
    g1[3] = 0xffff                                          // tensor_dim1 hi16
          | (TK << 16);                                     // tile_dim0 = 32
    g1[4] = tile_rows & 0xffff;                             // tile_dim1
    g1[5] = stride_elems;                                   // dim0_stride lo32
    g1[6] = 0;                                              // dim0_stride hi16
    g1[7] = 0;
    tdm_v4i gz4 = {0, 0, 0, 0};
    tdm_v8i gz8 = {0, 0, 0, 0, 0, 0, 0, 0};
    __builtin_amdgcn_tensor_load_to_lds(g0, g1, gz4, gz4, gz8, 0);
}
#endif

// ---------------------------------------------------------------------------
// bf16-WMMA GEMM:  C[rows, NC] = epi( A[rows, K] @ W[K, NC] + bias )
// A is bf16 [rows,K]; Wt is bf16 PRE-TRANSPOSED [NC,K] (+ expert stride).
//   LEAKY    : LeakyReLU(0.01) epilogue
//   GATHER_A : A row index via perm[] (routed decoder layer 1)
//   SCATTER_C: C row index via perm[] (routed decoder layer 2)
//   ROUTED   : per-block expert selection via padded offsets[]
//   OUT_BF16 : store C as bf16 (intermediate activations) else fp32
// Non-gathered kernels stage tiles with the TDM (1 instr/tile); the gathered
// kernel uses per-thread async b128 copies. Double-buffered, last-iter peeled.
// ---------------------------------------------------------------------------
template<bool LEAKY, bool GATHER_A, bool SCATTER_C, bool ROUTED, bool OUT_BF16>
__global__ __launch_bounds__(256)
void gemm_bf16_wmma(const __bf16* __restrict__ A,
                    const __bf16* __restrict__ Wt,
                    const float*  __restrict__ bg,
                    void* __restrict__ Cout,
                    int K, int NC,
                    const int* __restrict__ perm,
                    const int* __restrict__ offsets,
                    long wstride, long bstride)
{
    constexpr bool USE_TDM = (HAVE_TDM != 0) && !GATHER_A;

    __shared__ __bf16 sA[2][TM][TK];    // 2 x 8 KB
    __shared__ __bf16 sB[2][TNB][TK];   // 2 x 8 KB, Wt tile: [n][k]

    const int tid  = threadIdx.x;
    const int row0 = blockIdx.x * TM;
    const int col0 = blockIdx.y * TNB;

    const __bf16* Wsel = Wt;
    const float*  bias = bg;
    if (ROUTED) {
        const int total = offsets[8];          // padded compact row count
        if (row0 >= total) return;             // uniform per block
        int e = 0;
        while (e < 7 && row0 >= offsets[e + 1]) ++e;
        Wsel = Wt + (long)e * wstride;
        bias = bg + (long)e * bstride;
    }

    const unsigned long abase = (unsigned long)A;
    const unsigned long bbase = (unsigned long)Wsel;
    const unsigned ldsA0 = (unsigned)(size_t)&sA[0][0][0];
    const unsigned ldsB0 = (unsigned)(size_t)&sB[0][0][0];
    const unsigned szA = (unsigned)(TM  * TK * 2);   // bytes per buffer
    const unsigned szB = (unsigned)(TNB * TK * 2);

    // ---- per-thread async fallback assignment (16B chunks, 4/thread) ----
    const int ar0 = tid >> 2;                 // rows 0..63
    const int ar1 = ar0 + 64;                 // rows 64..127
    const int akc = (tid & 3) * 8;            // k-chunk (8 bf16 = 16B)
    unsigned avo0 = 0, avo1 = 0, bvo0 = 0, bvo1 = 0;
    unsigned lA0 = 0, lA1 = 0, lB0 = 0, lB1 = 0;
    if (!USE_TDM) {
        long arow0, arow1;
        if (GATHER_A) {
            const int p0 = perm[row0 + ar0]; arow0 = (p0 < 0) ? 0 : p0;
            const int p1 = perm[row0 + ar1]; arow1 = (p1 < 0) ? 0 : p1;
        } else { arow0 = row0 + ar0; arow1 = row0 + ar1; }
        avo0 = (unsigned)((arow0 * (long)K + akc) * 2);
        avo1 = (unsigned)((arow1 * (long)K + akc) * 2);
        bvo0 = (unsigned)(((long)(col0 + ar0) * K + akc) * 2);
        bvo1 = (unsigned)(((long)(col0 + ar1) * K + akc) * 2);
        lA0 = ldsA0 + (unsigned)(ar0 * TK + akc) * 2;
        lA1 = ldsA0 + (unsigned)(ar1 * TK + akc) * 2;
        lB0 = ldsB0 + (unsigned)(ar0 * TK + akc) * 2;
        lB1 = ldsB0 + (unsigned)(ar1 * TK + akc) * 2;
    }

    // ---- wave / fragment coordinates (2x4 wave grid, 64x32 per wave) ----
    const int lane = tid & 31;
    const int wid  = tid >> 5;
    const int wmb  = (wid & 1) * 64;
    const int wnb  = (wid >> 1) * 32;
    const int half = lane >> 4;
    const int lr   = lane & 15;

    v8f acc[4][2] = {};

    auto stage = [&](int bi, int kbyte) {
        if (USE_TDM) {
#if HAVE_TDM
            if (wid == 0) {
                tdm_load_2d(ldsA0 + (unsigned)bi * szA,
                            abase + (unsigned long)((long)row0 * K * 2 + kbyte),
                            TM, K);
                tdm_load_2d(ldsB0 + (unsigned)bi * szB,
                            bbase + (unsigned long)((long)col0 * K * 2 + kbyte),
                            TNB, K);
            }
#endif
        } else {
            const unsigned ao = (unsigned)bi * szA, bo = (unsigned)bi * szB;
            async_cp_b128(lA0 + ao, avo0 + kbyte, abase);
            async_cp_b128(lA1 + ao, avo1 + kbyte, abase);
            async_cp_b128(lB0 + bo, bvo0 + kbyte, bbase);
            async_cp_b128(lB1 + bo, bvo1 + kbyte, bbase);
        }
    };
    auto wait_prev = [&]() {   // previous stage done, next still in flight
        if (USE_TDM) {
#if HAVE_TDM
            if (wid == 0) __builtin_amdgcn_s_wait_tensorcnt(2);
#endif
        } else {
            asm volatile("s_wait_asynccnt 0x4" ::: "memory");
        }
    };
    auto wait_zero = [&]() {
        if (USE_TDM) {
#if HAVE_TDM
            if (wid == 0) __builtin_amdgcn_s_wait_tensorcnt(0);
#endif
        } else {
            asm volatile("s_wait_asynccnt 0x0" ::: "memory");
        }
    };
    auto compute = [&](int bi) {
        // ISA 16-bit A layout: lanes 0-15 K{0..7,16..23}, 16-31 K{8..15,24..31}
        // B (Wt rows): same K order as A: lanes 0-15 K{0..7,16..23}, 16-31 K{8..15,24..31}
        v16bf afrag[4], bfrag[2];
        #pragma unroll
        for (int i = 0; i < 4; ++i) {
            const int r = wmb + i * 16 + lr;
            union { v16bf v; v8bf h[2]; } u;
            u.h[0] = *(const v8bf*)&sA[bi][r][half * 8];
            u.h[1] = *(const v8bf*)&sA[bi][r][half * 8 + 16];
            afrag[i] = u.v;
        }
        #pragma unroll
        for (int j = 0; j < 2; ++j) {
            const int c = wnb + j * 16 + lr;
            union { v16bf v; v8bf h[2]; } u;
            u.h[0] = *(const v8bf*)&sB[bi][c][half * 8];
            u.h[1] = *(const v8bf*)&sB[bi][c][half * 8 + 16];
            bfrag[j] = u.v;
        }
        #pragma unroll
        for (int i = 0; i < 4; ++i)
            #pragma unroll
            for (int j = 0; j < 2; ++j)
                acc[i][j] = __builtin_amdgcn_wmma_f32_16x16x32_bf16(
                    false, afrag[i], false, bfrag[j],
                    (short)0, acc[i][j], false, false);
    };

    // ---- software pipeline: double buffer, last iteration peeled ----
    stage(0, 0);
    int buf = 0;
    for (int k0 = 0; k0 < K - TK; k0 += TK) {
        stage(buf ^ 1, (k0 + TK) * 2);
        wait_prev();
        __syncthreads();
        compute(buf);
        __syncthreads();
        buf ^= 1;
    }
    wait_zero();
    __syncthreads();
    compute(buf);

    // ---- epilogue: bias + optional LeakyReLU ----
    // C frag layout: n = lane%16, m = vgpr + (lane>=16)*8
    float*  Cf = (float*)Cout;
    __bf16* Cb = (__bf16*)Cout;
    #pragma unroll
    for (int i = 0; i < 4; ++i) {
        #pragma unroll
        for (int j = 0; j < 2; ++j) {
            const int n = col0 + wnb + j * 16 + lr;
            const float b = bias[n];
            #pragma unroll
            for (int v = 0; v < 8; ++v) {
                const int m = row0 + wmb + i * 16 + half * 8 + v;
                float x = acc[i][j][v] + b;
                if (LEAKY) x = (x > 0.f) ? x : 0.01f * x;
                long orow;
                if (SCATTER_C) {
                    const int p = perm[m];
                    if (p < 0) continue;           // padding row
                    orow = p;
                } else {
                    orow = m;
                }
                if (OUT_BF16) Cb[orow * (long)NC + n] = (__bf16)x;
                else          Cf[orow * (long)NC + n] = x;
            }
        }
    }
}

// ---------------------------------------------------------------------------
// Preprocessing: fp32 -> bf16 conversion (plain and transposed)
// ---------------------------------------------------------------------------
__global__ void k_f32_to_bf16(const float* __restrict__ x, __bf16* __restrict__ y,
                              long n) {
    const long i = ((long)blockIdx.x * blockDim.x + threadIdx.x) * 4;
    if (i < n) {
        float4 f = *(const float4*)(x + i);
        y[i + 0] = (__bf16)f.x; y[i + 1] = (__bf16)f.y;
        y[i + 2] = (__bf16)f.z; y[i + 3] = (__bf16)f.w;
    }
}

// Wt[b][n][k] = (bf16) W[b][k][n];   grid=(NC/32, K/32, batch), block=(32,8)
__global__ void k_transpose_bf16(const float* __restrict__ W,
                                 __bf16* __restrict__ Wt, int K, int NC) {
    __shared__ float tile[32][33];
    const long bo = (long)blockIdx.z * K * NC;
    const int k0 = blockIdx.y * 32, n0 = blockIdx.x * 32;
    const int tx = threadIdx.x, ty = threadIdx.y;
    #pragma unroll
    for (int i = 0; i < 32; i += 8)
        tile[ty + i][tx] = W[bo + (long)(k0 + ty + i) * NC + n0 + tx];
    __syncthreads();
    #pragma unroll
    for (int i = 0; i < 32; i += 8)
        Wt[bo + (long)(n0 + ty + i) * K + k0 + tx] = (__bf16)tile[tx][ty + i];
}

// ---------------------------------------------------------------------------
// Routing + elementwise kernels
// ---------------------------------------------------------------------------
__global__ void k_zero_counts(int* cnt) { if (threadIdx.x < PS) cnt[threadIdx.x] = 0; }

__global__ void k_hist(const int* __restrict__ s, int* __restrict__ cnt, int n) {
    const int i = blockIdx.x * blockDim.x + threadIdx.x;
    if (i < n) atomicAdd(&cnt[s[i]], 1);
}

__global__ void k_scan_pad(const int* __restrict__ cnt, int* __restrict__ offs,
                           int* __restrict__ cur) {
    int off = 0;
    for (int e = 0; e < PS; ++e) {
        offs[e] = off;
        cur[e]  = off;
        off += ((cnt[e] + TM - 1) / TM) * TM;   // pad segments to 128 rows
    }
    offs[PS] = off;
}

__global__ void k_fill_perm(int* __restrict__ perm, int cap) {
    const int i = blockIdx.x * blockDim.x + threadIdx.x;
    if (i < cap) perm[i] = -1;
}

__global__ void k_scatter(const int* __restrict__ s, int* __restrict__ cur,
                          int* __restrict__ perm, int n) {
    const int i = blockIdx.x * blockDim.x + threadIdx.x;
    if (i < n) {
        const int e   = s[i];
        const int idx = atomicAdd(&cur[e], 1);
        perm[idx] = i;
    }
}

__global__ void k_hsample(const float* __restrict__ eps, const float* __restrict__ mu,
                          const float* __restrict__ lv, float* __restrict__ out,
                          __bf16* __restrict__ outb, long n) {
    const long i = (long)blockIdx.x * blockDim.x + threadIdx.x;
    if (i < n) {
        const float v = eps[i] * __expf(0.5f * lv[i]) + mu[i];
        out[i]  = v;
        outb[i] = (__bf16)v;
    }
}

// ---------------------------------------------------------------------------
// Launcher
// ---------------------------------------------------------------------------
extern "C" void kernel_launch(void* const* d_in, const int* in_sizes, int n_in,
                              void* d_out, int out_size, void* d_ws, size_t ws_size,
                              hipStream_t stream) {
    const float* data = (const float*)d_in[0];
    const float* eps  = (const float*)d_in[1];
    const float* Wm1  = (const float*)d_in[2];
    const float* bm1  = (const float*)d_in[3];
    const float* Wm2  = (const float*)d_in[4];
    const float* bm2  = (const float*)d_in[5];
    const float* Wv1  = (const float*)d_in[6];
    const float* bv1  = (const float*)d_in[7];
    const float* Wv2  = (const float*)d_in[8];
    const float* bv2  = (const float*)d_in[9];
    const float* We1  = (const float*)d_in[10];
    const float* be1  = (const float*)d_in[11];
    const float* We2  = (const float*)d_in[12];
    const float* be2  = (const float*)d_in[13];
    const int*   sg   = (const int*)d_in[14];

    // d_out: reconstruct[N,DX] | mu[N,DH] | logvar[N,DH] | h_sample[N,DH]
    float* out_rec = (float*)d_out;
    float* out_mu  = out_rec + (long)PN * PDX;
    float* out_lv  = out_mu  + (long)PN * PDH;
    float* out_h   = out_lv  + (long)PN * PDH;

    // workspace: bf16 operand copies + routing ints
    __bf16* data_bf = (__bf16*)d_ws;                     // [N, DX]
    __bf16* hs_bf   = data_bf + (long)PN * PDX;          // [N, DH]
    __bf16* hid_bf  = hs_bf   + (long)PN * PDH;          // [CAP_ROWS, DH]
    __bf16* Wm1t    = hid_bf  + (long)CAP_ROWS * PDH;    // [DH, DX]
    __bf16* Wm2t    = Wm1t + (long)PDX * PDH;            // [DH, DH]
    __bf16* Wv1t    = Wm2t + (long)PDH * PDH;
    __bf16* Wv2t    = Wv1t + (long)PDX * PDH;
    __bf16* We1t    = Wv2t + (long)PDH * PDH;            // [S, DH, DH]
    __bf16* We2t    = We1t + (long)PS * PDH * PDH;       // [S, DX, DH]
    int*    ints    = (int*)(We2t + (long)PS * PDH * PDX);
    int* cnt  = ints;            // 8
    int* offs = ints + 8;        // 9
    int* cur  = ints + 17;       // 8
    int* perm = ints + 25;       // CAP_ROWS

    // ---- one-time fp32 -> bf16 conversions (weights pre-transposed) ----
    k_f32_to_bf16<<<((long)PN * PDX / 4 + 255) / 256, 256, 0, stream>>>(
        data, data_bf, (long)PN * PDX);
    const dim3 tb(32, 8);
    k_transpose_bf16<<<dim3(PDH / 32, PDX / 32, 1), tb, 0, stream>>>(Wm1, Wm1t, PDX, PDH);
    k_transpose_bf16<<<dim3(PDH / 32, PDH / 32, 1), tb, 0, stream>>>(Wm2, Wm2t, PDH, PDH);
    k_transpose_bf16<<<dim3(PDH / 32, PDX / 32, 1), tb, 0, stream>>>(Wv1, Wv1t, PDX, PDH);
    k_transpose_bf16<<<dim3(PDH / 32, PDH / 32, 1), tb, 0, stream>>>(Wv2, Wv2t, PDH, PDH);
    k_transpose_bf16<<<dim3(PDH / 32, PDH / 32, PS), tb, 0, stream>>>(We1, We1t, PDH, PDH);
    k_transpose_bf16<<<dim3(PDX / 32, PDH / 32, PS), tb, 0, stream>>>(We2, We2t, PDH, PDX);

    // ---- routing: histogram -> padded scan -> scatter permutation ----
    k_zero_counts<<<1, 32, 0, stream>>>(cnt);
    k_hist<<<(PN + 255) / 256, 256, 0, stream>>>(sg, cnt, PN);
    k_scan_pad<<<1, 1, 0, stream>>>(cnt, offs, cur);
    k_fill_perm<<<(CAP_ROWS + 255) / 256, 256, 0, stream>>>(perm, CAP_ROWS);
    k_scatter<<<(PN + 255) / 256, 256, 0, stream>>>(sg, cur, perm, PN);

    const dim3 blk(256);
    const dim3 genc(PN / TM, PDH / TNB);      // 128 x 8 blocks

    // ---- encoder: mu_h ----
    gemm_bf16_wmma<true , false, false, false, true ><<<genc, blk, 0, stream>>>(
        data_bf, Wm1t, bm1, hid_bf, PDX, PDH, nullptr, nullptr, 0, 0);
    gemm_bf16_wmma<false, false, false, false, false><<<genc, blk, 0, stream>>>(
        hid_bf, Wm2t, bm2, out_mu, PDH, PDH, nullptr, nullptr, 0, 0);
    // ---- encoder: logvar_h ----
    gemm_bf16_wmma<true , false, false, false, true ><<<genc, blk, 0, stream>>>(
        data_bf, Wv1t, bv1, hid_bf, PDX, PDH, nullptr, nullptr, 0, 0);
    gemm_bf16_wmma<false, false, false, false, false><<<genc, blk, 0, stream>>>(
        hid_bf, Wv2t, bv2, out_lv, PDH, PDH, nullptr, nullptr, 0, 0);

    // ---- reparameterize (fp32 out + bf16 copy for decoder GEMM) ----
    k_hsample<<<((long)PN * PDH + 255) / 256, 256, 0, stream>>>(
        eps, out_mu, out_lv, out_h, hs_bf, (long)PN * PDH);

    // ---- routed decoder (only the selected expert per row) ----
    const dim3 gdec1(CAP_ROWS / TM, PDH / TNB);  // 136 x 8 (late blocks exit)
    gemm_bf16_wmma<true , true , false, true , true ><<<gdec1, blk, 0, stream>>>(
        hs_bf, We1t, be1, hid_bf, PDH, PDH, perm, offs, (long)PDH * PDH, PDH);
    const dim3 gdec2(CAP_ROWS / TM, PDX / TNB);
    gemm_bf16_wmma<false, false, true , true , false><<<gdec2, blk, 0, stream>>>(
        hid_bf, We2t, be2, out_rec, PDH, PDX, perm, offs, (long)PDH * PDX, PDX);
}
